// TF_52218212385532
// MI455X (gfx1250) — matrix operand order repaired and checked by reading, not compiled, not run on hardware
//
#include <hip/hip_runtime.h>
#include <hip/hip_bf16.h>
#include <cstdint>

// ---------------------------------------------------------------------------
// MI455X (gfx1250, wave32) implementation of the graph-transformer reference.
//  - All dense matmuls via v_wmma_f32_16x16x32_bf16 (bf16 operands, f32 acc)
//  - LDS tiles stored in WMMA *fragment order* so each operand fragment is one
//    contiguous 32B run -> v16bf load -> 2x ds_load_b128 (no scalar gathers)
//  - fp32->bf16 via __builtin_convertvector (native v_cvt_pk_bf16_f32 path)
//  - Tile loaders are branch-free: kernel requires K%32==0 && Nc%64==0
//    (all call sites satisfy this); only M is ragged (clamp + masked epilogue)
//  - Edge attention: 3 streaming passes, e = ea@We recomputed from LDS tile
//  - Segment max via ordered-uint atomicMax; segment sum via f32 atomicAdd
// ---------------------------------------------------------------------------

typedef __attribute__((ext_vector_type(16))) __bf16 v16bf;
typedef __attribute__((ext_vector_type(8)))  __bf16 v8bf;
typedef __attribute__((ext_vector_type(2)))  __bf16 v2bf;
typedef __attribute__((ext_vector_type(8)))  float  v8f;
typedef __attribute__((ext_vector_type(2)))  float  v2f;

static __device__ __forceinline__ v8bf cvt8(float4 lo, float4 hi) {
    v8f f = { lo.x, lo.y, lo.z, lo.w, hi.x, hi.y, hi.z, hi.w };
    return __builtin_convertvector(f, v8bf);      // packed cvt_pk_bf16_f32
}
static __device__ __forceinline__ unsigned cvt2_pack(float a, float b) {
    v2f f = { a, b };
    v2bf r = __builtin_convertvector(f, v2bf);
    return __builtin_bit_cast(unsigned, r);
}

static __device__ __forceinline__ unsigned f2ord(float f) {
    unsigned u = __float_as_uint(f);
    return (u & 0x80000000u) ? ~u : (u | 0x80000000u);   // monotonic mapping
}
static __device__ __forceinline__ float ord2f(unsigned e) {
    unsigned u = (e & 0x80000000u) ? (e & 0x7FFFFFFFu) : ~e;
    return __uint_as_float(u);
}

// ---------------------------------------------------------------------------
// WMMA GEMM: Out[M,Nc] = act(A[M,K] @ W[K,Nc] + bias)
// REQUIRES: K % 32 == 0, Nc % 64 == 0 (branch-free tile loads); M may be ragged.
// 64x64 tile per 128-thread workgroup; 4 waves x (16 rows x 4 wmma tiles).
// act: 0 = identity, 1 = leaky_relu(0.01)
//
// LDS layouts (bf16, 64B row stride, 32B-aligned fragment runs):
//   sA[r][k], k contiguous (8-wide K blocks stored in order, K0-31)
//     -> lane (m=l16, half) fragment = 32 contiguous bytes at sA[m][half*16]
//   sB2[n][k]  (B transposed; k contiguous)
//     -> lane (n=cb*16+l16, half) fragment = 32 bytes at sB2[n][half*16]
// ---------------------------------------------------------------------------
#define BM 64
#define BN 64
#define BK 32

__global__ __launch_bounds__(128)
void gemm_bf16_wmma_kernel(const float* __restrict__ A,
                           const float* __restrict__ W,
                           const float* __restrict__ bias,
                           float* __restrict__ Out,
                           int M, int K, int Nc, int act)
{
    __shared__ alignas(32) __bf16 sA [BM][BK];   // 4 KB
    __shared__ alignas(32) __bf16 sB2[BN][BK];   // 4 KB (transposed W tile)

    const int tile_m = blockIdx.x * BM;
    const int tile_n = blockIdx.y * BN;
    const int tid    = threadIdx.x;
    const int wave   = tid >> 5;          // 0..3 (wave32)
    const int lane   = tid & 31;
    const int half   = lane >> 4;         // 0: lanes 0-15, 1: lanes 16-31
    const int l16    = lane & 15;

    // cooperative-load index precompute
    const int a_r  = tid >> 1;            // A tile row 0..63
    const int a_b0 = (tid & 1) * 2;       // first of two 8-wide K blocks
    const int b_kp = tid >> 3;            // B tile k-pair 0..15 (k = 2*kp,2*kp+1)
    const int b_c0 = (tid & 7) * 8;       // B tile col group (8 cols)

    // A row clamped: OOB rows load duplicate data, epilogue masks the stores.
    const int a_gr = tile_m + a_r;
    const float* __restrict__ Arow =
        A + (size_t)((a_gr < M) ? a_gr : (M - 1)) * K;
    const int b_n0 = tile_n + b_c0;

    v8f acc[4] = {};

    const int nk = K / BK;                // K % 32 == 0 by contract
    for (int kt = 0; kt < nk; ++kt) {
        const int k0 = kt * BK;

        // ---- A tile: two v8bf (2x b128 load, packed cvt, b128 store) ----
#pragma unroll
        for (int bb = 0; bb < 2; ++bb) {
            const int b  = a_b0 + bb;            // K block 0..3
            const int kc = k0 + b * 8;
            float4 lo = *(const float4*)&Arow[kc];
            float4 hi = *(const float4*)&Arow[kc + 4];
            const int pb = b;                                // contiguous K block
            *(v8bf*)&sA[a_r][pb * 8] = cvt8(lo, hi);
        }

        // ---- B tile: row-pair of 8 cols (4x b128 load); store transposed ----
        {
            const int gk0 = k0 + 2 * b_kp;
            const float* W0 = &W[(size_t)gk0 * Nc + b_n0];
            const float* W1 = W0 + Nc;
            float4 a0 = *(const float4*)W0, a1 = *(const float4*)(W0 + 4);
            float4 b0 = *(const float4*)W1, b1 = *(const float4*)(W1 + 4);
            float r0[8] = { a0.x, a0.y, a0.z, a0.w, a1.x, a1.y, a1.z, a1.w };
            float r1[8] = { b0.x, b0.y, b0.z, b0.w, b1.x, b1.y, b1.z, b1.w };
#pragma unroll
            for (int j = 0; j < 8; ++j)
                *(unsigned*)&sB2[b_c0 + j][2 * b_kp] = cvt2_pack(r0[j], r1[j]);
        }
        // prefetch next A tile into cache hierarchy
        if (kt + 1 < nk)
            __builtin_prefetch(&Arow[k0 + BK + a_b0 * 8], 0, 1);
        __syncthreads();

        // ---- fragments: single contiguous 32B loads (2x ds_load_b128) ----
        const int arow = wave * 16 + l16;
        v16bf afrag = *(const v16bf*)&sA[arow][half * 16];
#pragma unroll
        for (int cb = 0; cb < 4; ++cb) {
            v16bf bfrag = *(const v16bf*)&sB2[cb * 16 + l16][half * 16];
            acc[cb] = __builtin_amdgcn_wmma_f32_16x16x32_bf16(
                false, afrag, false, bfrag, (short)0, acc[cb], false, false);
        }
        __syncthreads();
    }

    // epilogue: bias + activation + store (C/D layout: elem r -> M = r+8*half)
    const int row0 = tile_m + wave * 16;
    if ((tile_m + BM) <= M) {
        // interior tile: predicate-free stores
#pragma unroll
        for (int cb = 0; cb < 4; ++cb) {
            const int n = tile_n + cb * 16 + l16;
            const float bval = bias ? bias[n] : 0.f;
#pragma unroll
            for (int r = 0; r < 8; ++r) {
                const int m = row0 + r + half * 8;
                float o = acc[cb][r] + bval;
                if (act == 1) o = (o > 0.f) ? o : 0.01f * o;
                Out[(size_t)m * Nc + n] = o;
            }
        }
    } else {
#pragma unroll
        for (int cb = 0; cb < 4; ++cb) {
            const int n = tile_n + cb * 16 + l16;
            const float bval = bias ? bias[n] : 0.f;
#pragma unroll
            for (int r = 0; r < 8; ++r) {
                const int m = row0 + r + half * 8;
                if (m < M) {
                    float o = acc[cb][r] + bval;
                    if (act == 1) o = (o > 0.f) ? o : 0.01f * o;
                    Out[(size_t)m * Nc + n] = o;
                }
            }
        }
    }
}

// ---------------------------------------------------------------------------
// Edge pass A: alpha[e] = dot(q[dst], k[src] + ea[e]@We) / 8 ; segment max
// ---------------------------------------------------------------------------
__global__ __launch_bounds__(256)
void edge_alpha_kernel(const int* __restrict__ src, const int* __restrict__ dst,
                       const float* __restrict__ ea, const float* __restrict__ We,
                       const float* __restrict__ q, const float* __restrict__ k,
                       float* __restrict__ alpha, unsigned* __restrict__ menc,
                       int nE)
{
    __shared__ float sWe[16 * 64];
    for (int i = threadIdx.x; i < 16 * 64; i += 256) sWe[i] = We[i];
    __syncthreads();

    int e = blockIdx.x * 256 + threadIdx.x;
    if (e >= nE) return;
    int s = src[e], d = dst[e];

    float a[16];
    {
        const float4* ea4 = (const float4*)(ea + (size_t)e * 16);
#pragma unroll
        for (int jj = 0; jj < 4; ++jj) {
            float4 t = ea4[jj];
            a[jj * 4 + 0] = t.x; a[jj * 4 + 1] = t.y;
            a[jj * 4 + 2] = t.z; a[jj * 4 + 3] = t.w;
        }
    }

    const float4* q4 = (const float4*)(q + (size_t)d * 64);
    const float4* k4 = (const float4*)(k + (size_t)s * 64);
    float acc = 0.f;
    for (int cq = 0; cq < 16; ++cq) {
        float4 qv = q4[cq], kv = k4[cq];
        float qa[4] = { qv.x, qv.y, qv.z, qv.w };
        float ka[4] = { kv.x, kv.y, kv.z, kv.w };
#pragma unroll
        for (int u = 0; u < 4; ++u) {
            const int c = cq * 4 + u;
            float ec = 0.f;
#pragma unroll
            for (int j = 0; j < 16; ++j) ec = fmaf(a[j], sWe[j * 64 + c], ec);
            acc = fmaf(qa[u], ka[u] + ec, acc);
        }
    }
    acc *= 0.125f;                       // 1/sqrt(64)
    alpha[e] = acc;
    atomicMax(&menc[d], f2ord(acc));
}

// ---------------------------------------------------------------------------
// Edge pass B: p = exp(alpha - m[dst]) ; segment sum of p
// ---------------------------------------------------------------------------
__global__ __launch_bounds__(256)
void edge_softmax_kernel(const int* __restrict__ dst,
                         float* __restrict__ alpha,
                         const unsigned* __restrict__ menc,
                         float* __restrict__ ssum, int nE)
{
    int e = blockIdx.x * 256 + threadIdx.x;
    if (e >= nE) return;
    int d = dst[e];
    unsigned enc = menc[d];
    float m = ord2f(enc);
    if (!isfinite(m)) m = 0.f;           // PyG: where(isfinite(m), m, 0)
    float p = __expf(alpha[e] - m);
    alpha[e] = p;
    atomicAdd(&ssum[d], p);
}

// ---------------------------------------------------------------------------
// Edge pass C: out[dst] += (v[src] + ea@We) * p / (s[dst]+1e-16)
// one thread per (edge, channel)
// ---------------------------------------------------------------------------
__global__ __launch_bounds__(256)
void edge_scatter_kernel(const int* __restrict__ src, const int* __restrict__ dst,
                         const float* __restrict__ ea, const float* __restrict__ We,
                         const float* __restrict__ v, const float* __restrict__ alpha,
                         const float* __restrict__ ssum, float* __restrict__ out,
                         int nE)
{
    __shared__ float sWe[16 * 64];
    for (int i = threadIdx.x; i < 16 * 64; i += 256) sWe[i] = We[i];
    __syncthreads();

    long long gid = (long long)blockIdx.x * 256 + threadIdx.x;
    if (gid >= (long long)nE * 64) return;
    int e = (int)(gid >> 6);
    int c = (int)(gid & 63);
    int s = src[e], d = dst[e];

    float p = alpha[e] / (ssum[d] + 1e-16f);
    float ec = 0.f;
#pragma unroll
    for (int j = 0; j < 16; ++j)
        ec = fmaf(ea[(size_t)e * 16 + j], sWe[j * 64 + c], ec);
    float val = (v[(size_t)s * 64 + c] + ec) * p;
    atomicAdd(&out[(size_t)d * 64 + c], val);
}

// ---------------------------------------------------------------------------
// Elementwise leaky relu (in place)
// ---------------------------------------------------------------------------
__global__ __launch_bounds__(256)
void leaky_kernel(float* __restrict__ h, long long n)
{
    long long i = (long long)blockIdx.x * 256 + threadIdx.x;
    if (i < n) {
        float v = h[i];
        h[i] = (v > 0.f) ? v : 0.01f * v;
    }
}

// ---------------------------------------------------------------------------
// global_add_pool: pool[batch[node]] += h[node]   (thread per (node,channel))
// ---------------------------------------------------------------------------
__global__ __launch_bounds__(256)
void pool_kernel(const float* __restrict__ h, const int* __restrict__ batch,
                 float* __restrict__ pool, int n)
{
    long long gid = (long long)blockIdx.x * 256 + threadIdx.x;
    if (gid >= (long long)n * 64) return;
    int node = (int)(gid >> 6);
    int c    = (int)(gid & 63);
    atomicAdd(&pool[(size_t)batch[node] * 64 + c], h[(size_t)node * 64 + c]);
}

// ---------------------------------------------------------------------------
// concat[node] = [hs0 | hs1 | hs2 | pool[batch[node]]]   (N x 256)
// ---------------------------------------------------------------------------
__global__ __launch_bounds__(256)
void concat_kernel(const float* __restrict__ h0, const float* __restrict__ h1,
                   const float* __restrict__ h2, const float* __restrict__ pool,
                   const int* __restrict__ batch, float* __restrict__ outc, int n)
{
    long long gid = (long long)blockIdx.x * 256 + threadIdx.x;
    if (gid >= (long long)n * 256) return;
    int node = (int)(gid >> 8);
    int j    = (int)(gid & 255);
    float v;
    if      (j < 64)  v = h0[(size_t)node * 64 + j];
    else if (j < 128) v = h1[(size_t)node * 64 + (j - 64)];
    else if (j < 192) v = h2[(size_t)node * 64 + (j - 128)];
    else              v = pool[(size_t)batch[node] * 64 + (j - 192)];
    outc[gid] = v;
}

// ---------------------------------------------------------------------------
// final: out[node] = sigmoid(h[node,:] . Wf + bf)
// ---------------------------------------------------------------------------
__global__ __launch_bounds__(256)
void final_kernel(const float* __restrict__ h, const float* __restrict__ Wf,
                  const float* __restrict__ bf, float* __restrict__ out, int n)
{
    __shared__ float sW[256];
    for (int i = threadIdx.x; i < 256; i += 256) sW[i] = Wf[i];
    __syncthreads();

    int node = blockIdx.x * 256 + threadIdx.x;
    if (node >= n) return;
    const float4* row4 = (const float4*)(h + (size_t)node * 256);
    float acc = bf[0];
    for (int i = 0; i < 64; ++i) {
        float4 rv = row4[i];
        acc = fmaf(rv.x, sW[4 * i + 0], acc);
        acc = fmaf(rv.y, sW[4 * i + 1], acc);
        acc = fmaf(rv.z, sW[4 * i + 2], acc);
        acc = fmaf(rv.w, sW[4 * i + 3], acc);
    }
    out[node] = 1.f / (1.f + __expf(-acc));
}

// ---------------------------------------------------------------------------
// Host orchestration
// ---------------------------------------------------------------------------
extern "C" void kernel_launch(void* const* d_in, const int* in_sizes, int n_in,
                              void* d_out, int out_size, void* d_ws, size_t ws_size,
                              hipStream_t stream)
{
    (void)n_in; (void)out_size; (void)ws_size;

    const int IN = 128, C = 64, ED = 16, L = 3, SC = 256, NM = 4000;
    const int N = in_sizes[0] / IN;
    const int E = in_sizes[1] / 2;

    const float* x     = (const float*)d_in[0];
    const int*   eidx  = (const int*)  d_in[1];
    const float* ea    = (const float*)d_in[2];
    const int*   batch = (const int*)  d_in[3];
    const float* Wq1 = (const float*)d_in[4];  const float* bq1 = (const float*)d_in[5];
    const float* Wk1 = (const float*)d_in[6];  const float* bk1 = (const float*)d_in[7];
    const float* Wv1 = (const float*)d_in[8];  const float* bv1 = (const float*)d_in[9];
    const float* We1 = (const float*)d_in[10];
    const float* Ws1 = (const float*)d_in[11]; const float* bs1 = (const float*)d_in[12];
    const float* Wq  = (const float*)d_in[13]; const float* bq  = (const float*)d_in[14];
    const float* Wk  = (const float*)d_in[15]; const float* bk  = (const float*)d_in[16];
    const float* Wv  = (const float*)d_in[17]; const float* bv  = (const float*)d_in[18];
    const float* We  = (const float*)d_in[19];
    const float* Ws  = (const float*)d_in[20]; const float* bs  = (const float*)d_in[21];
    const float* W1  = (const float*)d_in[22]; const float* b1  = (const float*)d_in[23];
    const float* Wc  = (const float*)d_in[24]; const float* bc  = (const float*)d_in[25];
    const float* Wf  = (const float*)d_in[26]; const float* bfp = (const float*)d_in[27];

    const int* srcp = eidx;
    const int* dstp = eidx + E;

    // ---- workspace layout (floats) ----
    const size_t N64 = (size_t)N * C;
    float* ws    = (float*)d_ws;
    float* hA    = ws;                 // conv1 output / layer0 input
    float* hs0   = hA  + N64;
    float* hs1   = hs0 + N64;
    float* hs2   = hs1 + N64;
    float* qb    = hs2 + N64;
    float* kb    = qb  + N64;
    float* vb    = kb  + N64;
    float* alpha = vb  + N64;          // E
    unsigned* menc = (unsigned*)(alpha + E);   // N
    float* ssum  = (float*)(menc + N);         // N
    float* pool  = ssum + N;                   // NM*64
    float* cat   = pool + (size_t)NM * C;      // N*256
    float* c1    = cat  + (size_t)N * SC;      // N*256
    float* c2    = c1   + (size_t)N * SC;      // N*256

    auto gemm = [&](const float* A, const float* Wm, const float* bias,
                    float* Out, int M, int K, int Nc, int act) {
        // contract: K % 32 == 0, Nc % 64 == 0
        dim3 g((M + BM - 1) / BM, Nc / BN);
        gemm_bf16_wmma_kernel<<<g, 128, 0, stream>>>(A, Wm, bias, Out, M, K, Nc, act);
    };

    const int eb   = (E + 255) / 256;
    const int ecb  = (int)(((long long)E * 64 + 255) / 256);
    const int nb64 = (int)(((long long)N * 64 + 255) / 256);

    auto tconv = [&](const float* hin, int K,
                     const float* Wq_, const float* bq_, const float* Wk_, const float* bk_,
                     const float* Wv_, const float* bv_, const float* We_,
                     const float* Ws_, const float* bs_, float* hout) {
        gemm(hin, Wq_, bq_, qb,   N, K, C, 0);
        gemm(hin, Wk_, bk_, kb,   N, K, C, 0);
        gemm(hin, Wv_, bv_, vb,   N, K, C, 0);
        gemm(hin, Ws_, bs_, hout, N, K, C, 0);       // skip term accumulates in-place
        hipMemsetAsync(menc, 0, (size_t)N * 4, stream);   // ord(-inf) == 0
        hipMemsetAsync(ssum, 0, (size_t)N * 4, stream);
        edge_alpha_kernel  <<<eb,  256, 0, stream>>>(srcp, dstp, ea, We_, qb, kb, alpha, menc, E);
        edge_softmax_kernel<<<eb,  256, 0, stream>>>(dstp, alpha, menc, ssum, E);
        edge_scatter_kernel<<<ecb, 256, 0, stream>>>(srcp, dstp, ea, We_, vb, alpha, ssum, hout, E);
        leaky_kernel       <<<nb64,256, 0, stream>>>(hout, (long long)N * C);
    };

    // conv1: IN -> C
    tconv(x, IN, Wq1, bq1, Wk1, bk1, Wv1, bv1, We1, Ws1, bs1, hA);
    // stacked layers: C -> C
    float* houts[3] = { hs0, hs1, hs2 };
    const float* hin = hA;
    for (int i = 0; i < L; ++i) {
        tconv(hin, C,
              Wq + (size_t)i * C * C, bq + (size_t)i * C,
              Wk + (size_t)i * C * C, bk + (size_t)i * C,
              Wv + (size_t)i * C * C, bv + (size_t)i * C,
              We + (size_t)i * ED * C,
              Ws + (size_t)i * C * C, bs + (size_t)i * C,
              houts[i]);
        hin = houts[i];
    }

    // global add pool + concat
    hipMemsetAsync(pool, 0, (size_t)NM * C * 4, stream);
    pool_kernel  <<<nb64, 256, 0, stream>>>(hs2, batch, pool, N);
    concat_kernel<<<(int)(((long long)N * SC + 255) / 256), 256, 0, stream>>>(
        hs0, hs1, hs2, pool, batch, cat, N);

    // classifier head
    gemm(cat, W1, b1, c1, N, C * (L + 1), SC, 0);                         // no act
    gemm(c1, Wc,                 bc,      c2, N, SC, SC, 1);              // leaky
    gemm(c2, Wc + (size_t)SC*SC, bc + SC, c1, N, SC, SC, 1);              // leaky
    final_kernel<<<(N + 255) / 256, 256, 0, stream>>>(c1, Wf, bfp, (float*)d_out, N);
}
